// MultiScaleAttention_16037407884049
// MI455X (gfx1250) — matrix operand order repaired and checked by reading, not compiled, not run on hardware
//
#include <hip/hip_runtime.h>

// ---------------------------------------------------------------------------
// MultiScaleAttention (MViT pooled attention) for MI455X / gfx1250, wave32.
// Pipeline: QKV GEMM (WMMA f16) -> depthwise3x3+LN pools -> flash attention
// (WMMA f16, online softmax, TDM-staged K tiles) -> proj GEMM (WMMA f16).
// fp32 accumulate via v_wmma_f32_16x16x32_f16 everywhere.
// ---------------------------------------------------------------------------

typedef __attribute__((ext_vector_type(16))) _Float16 v16h;
typedef __attribute__((ext_vector_type(8)))  float    v8f;
typedef __attribute__((ext_vector_type(4)))  unsigned int v4u;
typedef __attribute__((ext_vector_type(8)))  int      v8i;
typedef __attribute__((ext_vector_type(4)))  int      v4i;

#define BATCH      32
#define NQ         3136      // 56*56
#define NK         784       // 28*28
#define HW_IN      56
#define CH         96
#define EQKV       288
#define MTOT       (BATCH * NQ)   // 100352
#define KCHUNK     112            // 7 chunks of 112 keys = 784
#define KCHUNK_PAD 128            // zero-padded to multiple of 32 for PV WMMA
#define SCALE_ATTN 0.1020620726f  // 96^-0.5

#if defined(__has_builtin)
#if __has_builtin(__builtin_amdgcn_tensor_load_to_lds)
#define HAVE_TDM 1
#endif
#endif

union FragH { v16h v; unsigned int u[8]; };

__device__ __forceinline__ v8f wmma_f16(v16h a, v16h b, v8f c) {
  return __builtin_amdgcn_wmma_f32_16x16x32_f16(false, a, false, b, (short)0, c,
                                                false, false);
}

// wave-relative LDS byte offset of a __shared__ object (generic -> AS3 cast)
__device__ __forceinline__ unsigned lds_byte_off(const void* p) {
  return (unsigned)(unsigned long long)(
      (const __attribute__((address_space(3))) void*)p);
}

// A-matrix fragment (16x32 f16, this wave's row = lane%16).
// Layout: lanes 0-15 hold K {0..7,16..23}; lanes 16-31 hold K {8..15,24..31}.
__device__ __forceinline__ v16h load_a_frag(const _Float16* rowptr, int k0, int lane) {
  FragH f;
  const int h8 = (lane & 16) ? 8 : 0;
#pragma unroll
  for (int p = 0; p < 4; ++p) {
    f.u[p]     = *(const unsigned int*)(rowptr + k0 + h8 + 2 * p);
    f.u[p + 4] = *(const unsigned int*)(rowptr + k0 + 16 + h8 + 2 * p);
  }
  return f.v;
}

// B-matrix fragment (32x16 f16) read from a row-major B^T buffer:
// nrowptr points at B^T[n][0] for n = lane%16; lanes 0-15 hold K {0..7,16..23}, lanes 16-31 hold K {8..15,24..31}.
__device__ __forceinline__ v16h load_bt_frag(const _Float16* nrowptr, int k0, int lane) {
  FragH f;
  const int h8 = (lane & 16) ? 8 : 0;
#pragma unroll
  for (int p = 0; p < 4; ++p) {
    f.u[p]     = *(const unsigned int*)(nrowptr + k0 + h8 + 2 * p);
    f.u[p + 4] = *(const unsigned int*)(nrowptr + k0 + 16 + h8 + 2 * p);
  }
  return f.v;
}

// ---------------------------------------------------------------------------
// Kernel 1: qkv = x @ Wqkv^T + bqkv    (M=100352, K=96, N=288) -> fp32 q,k,v
// block = 128 threads (4 waves); block tile = 64 rows x 16 cols.
// ---------------------------------------------------------------------------
__global__ void __launch_bounds__(128)
k_qkv_gemm(const float* __restrict__ x, const float* __restrict__ Wqkv,
           const float* __restrict__ bqkv, float* __restrict__ qf,
           float* __restrict__ kf, float* __restrict__ vf) {
  __shared__ __align__(16) _Float16 As[64 * CH];   // 12 KB
  __shared__ __align__(16) _Float16 Bs[16 * CH];   //  3 KB

  const int tid  = threadIdx.x;
  const int lane = tid & 31;
  const int wv   = tid >> 5;
  const int m0   = blockIdx.x * 64;
  const int n0   = blockIdx.y * 16;

  // stage A tile (64x96) fp32 -> f16
  for (int i = tid; i < 64 * CH; i += 128) {
    __builtin_prefetch(&x[(size_t)(m0 + (i >> 6)) * CH], 0, 0);
    As[i] = (_Float16)x[(size_t)(m0 + i / CH) * CH + (i % CH)];
  }
  // stage B tile: rows n0..n0+15 of Wqkv (row-major == B^T layout)
  for (int i = tid; i < 16 * CH; i += 128)
    Bs[i] = (_Float16)Wqkv[(size_t)(n0 + i / CH) * CH + (i % CH)];
  __syncthreads();

  const _Float16* arow = As + (wv * 16 + (lane & 15)) * CH;
  const _Float16* brow = Bs + (lane & 15) * CH;
  v8f acc = {};
#pragma unroll
  for (int ks = 0; ks < 3; ++ks)
    acc = wmma_f16(load_a_frag(arow, ks * 32, lane),
                   load_bt_frag(brow, ks * 32, lane), acc);

  const int e    = n0 + (lane & 15);
  const float bb = bqkv[e];
  const int buf  = e / CH;
  const int ch   = e - buf * CH;
  float* dst = (buf == 0) ? qf : (buf == 1) ? kf : vf;
  const int hh = (lane & 16) ? 8 : 0;
#pragma unroll
  for (int r = 0; r < 8; ++r) {
    const int m = m0 + wv * 16 + r + hh;
    dst[(size_t)m * CH + ch] = acc[r] + bb;
  }
}

// ---------------------------------------------------------------------------
// Kernel 2: depthwise 3x3 conv (pad 1) + LayerNorm over channels.
// One wave32 per output token; 3 channels per lane (96 = 32*3).
// ---------------------------------------------------------------------------
__global__ void __launch_bounds__(128)
k_pool_ln(const float* __restrict__ in, const float* __restrict__ w9,
          const float* __restrict__ g, const float* __restrict__ beta,
          _Float16* __restrict__ out_h, float* __restrict__ out_f,
          int stride, int Hp, int Wp) {
  const int lane  = threadIdx.x & 31;
  const int token = blockIdx.x * 4 + (threadIdx.x >> 5);
  const int Lp    = Hp * Wp;
  if (token >= BATCH * Lp) return;
  const int b  = token / Lp;
  const int t  = token - b * Lp;
  const int oy = t / Wp;
  const int ox = t - oy * Wp;

  float wr[3][9];
#pragma unroll
  for (int u = 0; u < 3; ++u)
#pragma unroll
    for (int k = 0; k < 9; ++k) wr[u][k] = w9[(lane + 32 * u) * 9 + k];

  float acc[3] = {0.f, 0.f, 0.f};
#pragma unroll
  for (int dy = 0; dy < 3; ++dy) {
    const int iy = oy * stride + dy - 1;
    if (iy < 0 || iy >= HW_IN) continue;
#pragma unroll
    for (int dx = 0; dx < 3; ++dx) {
      const int ix = ox * stride + dx - 1;
      if (ix < 0 || ix >= HW_IN) continue;
      const size_t base = ((size_t)b * NQ + iy * HW_IN + ix) * CH;
#pragma unroll
      for (int u = 0; u < 3; ++u)
        acc[u] += in[base + lane + 32 * u] * wr[u][dy * 3 + dx];
    }
  }

  // LayerNorm over 96 channels (full-wave reduction, wave32)
  float s1 = acc[0] + acc[1] + acc[2];
  float s2 = acc[0] * acc[0] + acc[1] * acc[1] + acc[2] * acc[2];
#pragma unroll
  for (int off = 1; off < 32; off <<= 1) {
    s1 += __shfl_xor(s1, off);
    s2 += __shfl_xor(s2, off);
  }
  const float mean = s1 * (1.0f / CH);
  const float var  = s2 * (1.0f / CH) - mean * mean;
  const float rs   = rsqrtf(var + 1e-6f);
#pragma unroll
  for (int u = 0; u < 3; ++u) {
    const int c   = lane + 32 * u;
    const float y = (acc[u] - mean) * rs * g[c] + beta[c];
    const size_t idx = (size_t)token * CH + c;
    out_h[idx] = (_Float16)y;
    if (out_f) out_f[idx] = y;
  }
}

// ---------------------------------------------------------------------------
// Kernel 3: flash attention. grid (49, 32), block 128 (4 waves).
// Each wave: 16 q rows. Keys streamed in 7 chunks of 112 through LDS;
// the contiguous K tile (112x96 f16) is DMA'd by the Tensor Data Mover.
// ---------------------------------------------------------------------------
__global__ void __launch_bounds__(128)
k_attn(const _Float16* __restrict__ qh, const _Float16* __restrict__ kh,
       const _Float16* __restrict__ vh, const float* __restrict__ qf,
       float* __restrict__ of) {
  __shared__ __align__(16) _Float16 smem[KCHUNK * CH + CH * KCHUNK_PAD +
                                         4 * 16 * KCHUNK_PAD];  // 62464 B
  _Float16* Ks = smem;                        // [112][96]  key-major
  _Float16* Vt = Ks + KCHUNK * CH;            // [96][128]  channel-major (B^T)
  _Float16* Ps = Vt + CH * KCHUNK_PAD;        // per-wave [16][128]

  const int tid   = threadIdx.x;
  const int lane  = tid & 31;
  const int wv    = tid >> 5;
  const int laneN = lane & 15;
  const int hh    = (lane & 16) ? 8 : 0;
  const int b     = blockIdx.y;
  const int qbase = blockIdx.x * 64 + wv * 16;

  // zero pad regions once: Vt keys 112..127, Ps cols 112..127
  for (int i = tid; i < CH * 16; i += 128)
    Vt[(i >> 4) * KCHUNK_PAD + KCHUNK + (i & 15)] = (_Float16)0.f;
  for (int i = tid; i < 4 * 16 * 16; i += 128) {
    const int w_ = i >> 8, rem = i & 255;
    Ps[w_ * 16 * KCHUNK_PAD + (rem >> 4) * KCHUNK_PAD + KCHUNK + (rem & 15)] =
        (_Float16)0.f;
  }

#ifdef HAVE_TDM
  // Tensor DMA descriptor pieces that are chunk-invariant (2D tile in a 2D
  // tensor): data_size=2B, tensor_dim0=96, tensor_dim1=784, tile=[96 x 112],
  // tensor_dim0_stride=96 elements.
  const v8i tdm_g1 = {(int)(1u << 16),          // data_size=2B, mask=0
                      (int)((unsigned)CH << 16),  // tensor_dim0 lo16
                      (int)((unsigned)NK << 16),  // dim0 hi=0 | tensor_dim1 lo16
                      (int)((unsigned)CH << 16),  // dim1 hi=0 | tile_dim0=96
                      (int)KCHUNK,                // tile_dim1=112 | tile_dim2=0
                      (int)CH,                    // tensor_dim0_stride lo32
                      0, 0};
  const v4i tdm_z4 = {0, 0, 0, 0};
  const unsigned ks_lds = lds_byte_off(Ks);
#endif

  // this wave's q tile (16x96) as three A fragments
  const _Float16* qrow = qh + ((size_t)b * NQ + qbase + laneN) * CH;
  v16h qa[3];
#pragma unroll
  for (int ks = 0; ks < 3; ++ks) qa[ks] = load_a_frag(qrow, ks * 32, lane);

  v8f oc[6];
#pragma unroll
  for (int t6 = 0; t6 < 6; ++t6) oc[t6] = (v8f){};
  float mr[8], lr[8];
#pragma unroll
  for (int r = 0; r < 8; ++r) { mr[r] = -1e30f; lr[r] = 0.f; }

  _Float16* Pw = Ps + wv * 16 * KCHUNK_PAD;

  for (int c = 0; c < 7; ++c) {
    const int kb = c * KCHUNK;
    __syncthreads();  // previous chunk fully consumed before reload

#ifdef HAVE_TDM
    // K chunk via Tensor Data Mover: global [112][96] f16 -> LDS Ks
    if (wv == 0) {
      const unsigned long long ga =
          (unsigned long long)(const void*)(kh + ((size_t)b * NK + kb) * CH);
      const v4u g0 = {1u,  // count=1 (valid user descriptor)
                      ks_lds, (unsigned)(ga & 0xffffffffu),
                      (unsigned)((ga >> 32) & 0x1ffffffu) | (2u << 30)};  // type=2
#if __clang_major__ >= 23
      const v8i z8 = {0, 0, 0, 0, 0, 0, 0, 0};
      __builtin_amdgcn_tensor_load_to_lds(g0, tdm_g1, tdm_z4, tdm_z4, z8, 0);
#else
      __builtin_amdgcn_tensor_load_to_lds(g0, tdm_g1, tdm_z4, tdm_z4, 0);
#endif
    }
#else
    // fallback: dword copy of contiguous K chunk
    {
      const unsigned int* ksrc =
          (const unsigned int*)(kh + ((size_t)b * NK + kb) * CH);
      unsigned int* kdst = (unsigned int*)Ks;
      for (int i = tid; i < KCHUNK * CH / 2; i += 128) kdst[i] = ksrc[i];
    }
#endif

    // V chunk transposed into Vt[chan][key] (overlaps with the TDM transfer)
    const _Float16* vsrc = vh + ((size_t)b * NK + kb) * CH;
    for (int i = tid; i < KCHUNK * CH; i += 128) {
      const int key = i / CH, chn = i - key * CH;
      Vt[chn * KCHUNK_PAD + key] = vsrc[(size_t)key * CH + chn];
    }

#ifdef HAVE_TDM
    if (wv == 0) __builtin_amdgcn_s_wait_tensorcnt(0);
#endif
    __syncthreads();

    // S = scale * q @ Kchunk^T : 7 tiles of 16x16
    v8f s[7];
#pragma unroll
    for (int nt = 0; nt < 7; ++nt) {
      v8f sc = {};
      const _Float16* krow = Ks + (nt * 16 + laneN) * CH;
#pragma unroll
      for (int ks = 0; ks < 3; ++ks)
        sc = wmma_f16(qa[ks], load_bt_frag(krow, ks * 32, lane), sc);
#pragma unroll
      for (int r = 0; r < 8; ++r) sc[r] *= SCALE_ATTN;
      s[nt] = sc;
    }

    // online softmax: row max (16-lane half == one D row), rescale, exp
    float alpha[8];
#pragma unroll
    for (int r = 0; r < 8; ++r) {
      float m_ = s[0][r];
#pragma unroll
      for (int nt = 1; nt < 7; ++nt) m_ = fmaxf(m_, s[nt][r]);
#pragma unroll
      for (int off = 1; off < 16; off <<= 1) m_ = fmaxf(m_, __shfl_xor(m_, off));
      const float mn = fmaxf(mr[r], m_);
      alpha[r] = __expf(mr[r] - mn);
      mr[r] = mn;
    }
    float rs[8];
#pragma unroll
    for (int r = 0; r < 8; ++r) rs[r] = 0.f;
#pragma unroll
    for (int nt = 0; nt < 7; ++nt)
#pragma unroll
      for (int r = 0; r < 8; ++r) {
        const float p = __expf(s[nt][r] - mr[r]);
        rs[r] += p;
        Pw[(r + hh) * KCHUNK_PAD + nt * 16 + laneN] = (_Float16)p;
      }
#pragma unroll
    for (int r = 0; r < 8; ++r) {
      float t_ = rs[r];
#pragma unroll
      for (int off = 1; off < 16; off <<= 1) t_ += __shfl_xor(t_, off);
      lr[r] = lr[r] * alpha[r] + t_;
    }
#pragma unroll
    for (int t6 = 0; t6 < 6; ++t6)
#pragma unroll
      for (int r = 0; r < 8; ++r) oc[t6][r] *= alpha[r];

    // O += P @ Vchunk  (K padded to 128 with zero P)
    const _Float16* prow = Pw + laneN * KCHUNK_PAD;
#pragma unroll
    for (int kk = 0; kk < 4; ++kk) {
      const v16h pa = load_a_frag(prow, kk * 32, lane);
#pragma unroll
      for (int t6 = 0; t6 < 6; ++t6) {
        const _Float16* vrow = Vt + (t6 * 16 + laneN) * KCHUNK_PAD;
        oc[t6] = wmma_f16(pa, load_bt_frag(vrow, kk * 32, lane), oc[t6]);
      }
    }
  }

  // epilogue: O/l + residual q (fp32), write fp32
#pragma unroll
  for (int t6 = 0; t6 < 6; ++t6) {
    const int chn = t6 * 16 + laneN;
#pragma unroll
    for (int r = 0; r < 8; ++r) {
      const int m = qbase + r + hh;
      const size_t idx = ((size_t)b * NQ + m) * CH + chn;
      of[idx] = oc[t6][r] / lr[r] + qf[idx];
    }
  }
}

// ---------------------------------------------------------------------------
// Kernel 4: out = attn_out @ Wproj^T + bproj  (M=100352, K=96, N=96)
// ---------------------------------------------------------------------------
__global__ void __launch_bounds__(128)
k_proj_gemm(const float* __restrict__ a, const float* __restrict__ Wproj,
            const float* __restrict__ bproj, float* __restrict__ out) {
  __shared__ __align__(16) _Float16 As[64 * CH];
  __shared__ __align__(16) _Float16 Bs[16 * CH];

  const int tid  = threadIdx.x;
  const int lane = tid & 31;
  const int wv   = tid >> 5;
  const int m0   = blockIdx.x * 64;
  const int n0   = blockIdx.y * 16;

  for (int i = tid; i < 64 * CH; i += 128)
    As[i] = (_Float16)a[(size_t)(m0 + i / CH) * CH + (i % CH)];
  for (int i = tid; i < 16 * CH; i += 128)
    Bs[i] = (_Float16)Wproj[(size_t)(n0 + i / CH) * CH + (i % CH)];
  __syncthreads();

  const _Float16* arow = As + (wv * 16 + (lane & 15)) * CH;
  const _Float16* brow = Bs + (lane & 15) * CH;
  v8f acc = {};
#pragma unroll
  for (int ks = 0; ks < 3; ++ks)
    acc = wmma_f16(load_a_frag(arow, ks * 32, lane),
                   load_bt_frag(brow, ks * 32, lane), acc);

  const int e    = n0 + (lane & 15);
  const float bb = bproj[e];
  const int hh   = (lane & 16) ? 8 : 0;
#pragma unroll
  for (int r = 0; r < 8; ++r) {
    const int m = m0 + wv * 16 + r + hh;
    out[(size_t)m * CH + e] = acc[r] + bb;
  }
}

// ---------------------------------------------------------------------------
extern "C" void kernel_launch(void* const* d_in, const int* in_sizes, int n_in,
                              void* d_out, int out_size, void* d_ws,
                              size_t ws_size, hipStream_t stream) {
  const float* x     = (const float*)d_in[0];
  const float* Wqkv  = (const float*)d_in[1];
  const float* bqkv  = (const float*)d_in[2];
  const float* pqw   = (const float*)d_in[3];
  const float* gq    = (const float*)d_in[4];
  const float* betq  = (const float*)d_in[5];
  const float* pkw   = (const float*)d_in[6];
  const float* gk    = (const float*)d_in[7];
  const float* betk  = (const float*)d_in[8];
  const float* pvw   = (const float*)d_in[9];
  const float* gv    = (const float*)d_in[10];
  const float* betv  = (const float*)d_in[11];
  const float* Wproj = (const float*)d_in[12];
  const float* bproj = (const float*)d_in[13];
  float* out = (float*)d_out;

  char* p = (char*)d_ws;
  float* qf = (float*)p;          p += (size_t)MTOT * CH * sizeof(float);
  float* kf = (float*)p;          p += (size_t)MTOT * CH * sizeof(float);
  float* vf = (float*)p;          p += (size_t)MTOT * CH * sizeof(float);
  _Float16* qp_h = (_Float16*)p;  p += (size_t)MTOT * CH * sizeof(_Float16);
  _Float16* kp_h = (_Float16*)p;  p += (size_t)BATCH * NK * CH * sizeof(_Float16);
  _Float16* vp_h = (_Float16*)p;  p += (size_t)BATCH * NK * CH * sizeof(_Float16);
  float* qp_f  = (float*)p;       p += (size_t)MTOT * CH * sizeof(float);
  float* att_f = (float*)p;       p += (size_t)MTOT * CH * sizeof(float);

  // 1) QKV projection
  k_qkv_gemm<<<dim3(MTOT / 64, EQKV / 16), 128, 0, stream>>>(x, Wqkv, bqkv, qf,
                                                             kf, vf);
  // 2-4) attention pools (conv3x3 depthwise + LN)
  k_pool_ln<<<(BATCH * NQ) / 4, 128, 0, stream>>>(qf, pqw, gq, betq, qp_h, qp_f,
                                                  1, 56, 56);
  k_pool_ln<<<(BATCH * NK) / 4, 128, 0, stream>>>(kf, pkw, gk, betk, kp_h,
                                                  nullptr, 2, 28, 28);
  k_pool_ln<<<(BATCH * NK) / 4, 128, 0, stream>>>(vf, pvw, gv, betv, vp_h,
                                                  nullptr, 2, 28, 28);
  // 5) flash attention + residual
  k_attn<<<dim3(NQ / 64, BATCH), 128, 0, stream>>>(qp_h, kp_h, vp_h, qp_f,
                                                   att_f);
  // 6) output projection
  k_proj_gemm<<<dim3(MTOT / 64, CH / 16), 128, 0, stream>>>(att_f, Wproj, bproj,
                                                            out);
}
